// MHA_6167573037151
// MI455X (gfx1250) — matrix operand order repaired and checked by reading, not compiled, not run on hardware
//
#include <hip/hip_runtime.h>

// B=2, S=4096, H=8, HD=64, D=512
#define S_LEN 4096
#define D_MODEL 512
#define NHEAD 8
#define HEADDIM 64
#define KV_STRIDE 72   // padded LDS row stride (bank-conflict avoidance)

typedef __attribute__((ext_vector_type(16))) __bf16 v16bf;
typedef __attribute__((ext_vector_type(8)))  __bf16 bf16x8;
typedef __attribute__((ext_vector_type(8)))  float  v8f;
typedef __attribute__((ext_vector_type(4)))  int    v4i;

// ---------------------------------------------------------------------------
// Async global -> LDS copy of 16 bytes (gfx1250 GLOBAL_LOAD_ASYNC_TO_LDS_B128),
// with a synchronous fallback if the builtin is unavailable.
// The builtin signature (from hipcc diagnostics) takes v4i pointers:
//   (addrspace(1) v4i*, addrspace(3) v4i*, imm offset, imm cpol)
// ---------------------------------------------------------------------------
__device__ __forceinline__ void async_copy16(const __bf16* g, __bf16* l) {
#if __has_builtin(__builtin_amdgcn_global_load_async_to_lds_b128)
  __builtin_amdgcn_global_load_async_to_lds_b128(
      (__attribute__((address_space(1))) v4i*)g,
      (__attribute__((address_space(3))) v4i*)l, 0, 0);
#else
  *(bf16x8*)l = *(const bf16x8*)g;
#endif
}

__device__ __forceinline__ void wait_async_all() {
#if __has_builtin(__builtin_amdgcn_global_load_async_to_lds_b128)
  asm volatile("s_wait_asynccnt 0" ::: "memory");
#endif
}

__device__ __forceinline__ v16bf lds_load32(const __bf16* p) {
  // 32 bytes as two 16B chunks (LDS row stride is only 16B-aligned)
  bf16x8 a = *(const bf16x8*)p;
  bf16x8 b = *(const bf16x8*)(p + 8);
  v16bf r;
#pragma unroll
  for (int e = 0; e < 8; ++e) { r[e] = a[e]; r[8 + e] = b[e]; }
  return r;
}

// ---------------------------------------------------------------------------
// Kernel 1: per-head QKV projection with WMMA (bf16 in, f32 acc, bf16 out)
// One wave handles one (16-token tile, head).  K,Q stored [B,H,S,64] bf16,
// V stored transposed [B,H,64,S] bf16.  Q is pre-scaled by 1/sqrt(64)=0.125
// (exact in bf16: exponent shift only).
// ---------------------------------------------------------------------------
template<bool TRANSPOSED>
__device__ __forceinline__ void proj_one(const v16bf a[2], const float* __restrict__ W,
                                         const float* __restrict__ bias,
                                         __bf16* __restrict__ out, int nn, int g,
                                         float outScale) {
#pragma unroll
  for (int nt = 0; nt < 4; ++nt) {
    v16bf wb[2];
#pragma unroll
    for (int kf = 0; kf < 2; ++kf)
#pragma unroll
      for (int e = 0; e < 16; ++e) {
        int k = kf * 32 + (e >> 3) * 16 + g * 8 + (e & 7);  // contraction index (d_in)
        wb[kf][e] = (__bf16)W[k * 64 + nt * 16 + nn];
      }
    float bv = bias[nt * 16 + nn];
    v8f acc;
#pragma unroll
    for (int r = 0; r < 8; ++r) acc[r] = bv;
    acc = __builtin_amdgcn_wmma_f32_16x16x32_bf16(false, a[0], false, wb[0], (short)0, acc, false, false);
    acc = __builtin_amdgcn_wmma_f32_16x16x32_bf16(false, a[1], false, wb[1], (short)0, acc, false, false);
#pragma unroll
    for (int r = 0; r < 8; ++r) {
      int row = g * 8 + r;                           // local token row 0..15
      float v = acc[r] * outScale;
      if (TRANSPOSED) out[(size_t)(nt * 16 + nn) * S_LEN + row] = (__bf16)v;
      else            out[(size_t)row * HEADDIM + nt * 16 + nn] = (__bf16)v;
    }
  }
}

__global__ __launch_bounds__(128) void qkv_proj_kernel(
    const float* __restrict__ x, const float* __restrict__ y,
    const float* __restrict__ Wk, const float* __restrict__ bk,
    const float* __restrict__ Wq, const float* __restrict__ bq,
    const float* __restrict__ Wv, const float* __restrict__ bv,
    __bf16* __restrict__ Qb, __bf16* __restrict__ Kb, __bf16* __restrict__ Vt) {
  const int lane = threadIdx.x & 31;
  const int wave = threadIdx.x >> 5;
  const int nn = lane & 15;
  const int g  = lane >> 4;
  const int task = blockIdx.x * 4 + wave;    // 0..4095
  const int mt = task >> 3;                  // token tile 0..511
  const int h  = task & 7;
  const int t0 = mt * 16;                    // first global token of tile
  const int b  = t0 >> 12;                   // t0 / 4096
  const int s0 = t0 & (S_LEN - 1);

  // A-fragments of x and y (16x32 bf16 each, two frags cover HD=64)
  v16bf ax[2], ay[2];
  {
    const float* xp = x + (size_t)(t0 + nn) * D_MODEL + h * HEADDIM;
    const float* yp = y + (size_t)(t0 + nn) * D_MODEL + h * HEADDIM;
#pragma unroll
    for (int f = 0; f < 2; ++f)
#pragma unroll
      for (int c = 0; c < 2; ++c) {
        const float* px = xp + f * 32 + c * 16 + g * 8;
        const float* py = yp + f * 32 + c * 16 + g * 8;
        float4 x0 = *(const float4*)(px);
        float4 x1 = *(const float4*)(px + 4);
        float4 y0 = *(const float4*)(py);
        float4 y1 = *(const float4*)(py + 4);
        ax[f][c*8+0]=(__bf16)x0.x; ax[f][c*8+1]=(__bf16)x0.y;
        ax[f][c*8+2]=(__bf16)x0.z; ax[f][c*8+3]=(__bf16)x0.w;
        ax[f][c*8+4]=(__bf16)x1.x; ax[f][c*8+5]=(__bf16)x1.y;
        ax[f][c*8+6]=(__bf16)x1.z; ax[f][c*8+7]=(__bf16)x1.w;
        ay[f][c*8+0]=(__bf16)y0.x; ay[f][c*8+1]=(__bf16)y0.y;
        ay[f][c*8+2]=(__bf16)y0.z; ay[f][c*8+3]=(__bf16)y0.w;
        ay[f][c*8+4]=(__bf16)y1.x; ay[f][c*8+5]=(__bf16)y1.y;
        ay[f][c*8+6]=(__bf16)y1.z; ay[f][c*8+7]=(__bf16)y1.w;
      }
  }

  const size_t bh = (size_t)b * NHEAD + h;
  __bf16* kout = Kb + (bh * S_LEN + s0) * HEADDIM;
  __bf16* qout = Qb + (bh * S_LEN + s0) * HEADDIM;
  __bf16* vout = Vt + bh * HEADDIM * S_LEN + s0;

  proj_one<false>(ax, Wk + h * 4096, bk + h * 64, kout, nn, g, 1.0f);    // K from x
  proj_one<false>(ay, Wq + h * 4096, bq + h * 64, qout, nn, g, 0.125f);  // Q from y (pre-scaled)
  proj_one<true >(ax, Wv + h * 4096, bv + h * 64, vout, nn, g, 1.0f);    // V from x (transposed)
}

// ---------------------------------------------------------------------------
// Kernel 2: flash attention.  Block = 4 waves sharing double-buffered K/V
// tiles in LDS (filled with async global->LDS DMA, ASYNCcnt-tracked), each
// wave owns a 16-query tile.  Online softmax, O in [B,S,H,64] f32.
// ---------------------------------------------------------------------------
__global__ __launch_bounds__(128) void flash_attn_kernel(
    const __bf16* __restrict__ Qb, const __bf16* __restrict__ Kb,
    const __bf16* __restrict__ Vt, float* __restrict__ O) {
  __shared__ __bf16 sK[2][64 * KV_STRIDE];   // K tile  [j'][d]   (18 KB)
  __shared__ __bf16 sV[2][64 * KV_STRIDE];   // V tile  [d][j']   (18 KB)
  __shared__ __bf16 sP[4][16 * KV_STRIDE];   // per-wave P bounce ( 9 KB)

  const int tid  = threadIdx.x;
  const int lane = tid & 31;
  const int wave = tid >> 5;
  const int nn = lane & 15;
  const int g  = lane >> 4;
  const int bh = blockIdx.y;                 // b*8 + h
  const int q0 = (blockIdx.x * 4 + wave) * 16;

  const __bf16* Qhead = Qb + (size_t)bh * S_LEN * HEADDIM;
  const __bf16* Khead = Kb + (size_t)bh * S_LEN * HEADDIM;
  const __bf16* Vhead = Vt + (size_t)bh * HEADDIM * S_LEN;
  __bf16* ldsP = sP[wave];

  // Q A-fragments (held in registers for the whole pass)
  v16bf qa[2];
  {
    const __bf16* qrow = Qhead + (size_t)(q0 + nn) * HEADDIM;
#pragma unroll
    for (int f = 0; f < 2; ++f)
#pragma unroll
      for (int c = 0; c < 2; ++c) {
        bf16x8 t = *(const bf16x8*)(qrow + f * 32 + c * 8 + g * 16);
#pragma unroll
        for (int e = 0; e < 8; ++e) qa[f][c * 8 + e] = t[e];
      }
  }

  float mrow[8], lrow[8];
  v8f o[4];
#pragma unroll
  for (int r = 0; r < 8; ++r) { mrow[r] = -1e30f; lrow[r] = 0.f; }
#pragma unroll
  for (int dt = 0; dt < 4; ++dt)
#pragma unroll
    for (int r = 0; r < 8; ++r) o[dt][r] = 0.f;

  // cooperative prefetch of one 64x64 K tile + V tile into LDS buffer `buf`
  auto prefetch = [&](int buf, int j0) {
#pragma unroll
    for (int i = 0; i < 4; ++i) {
      int chunk = i * 128 + tid;             // 0..511, 16B each
      int row = chunk >> 3;                  // 0..63
      int c8  = (chunk & 7) * 8;             // element offset 0..56
      async_copy16(Khead + (size_t)(j0 + row) * HEADDIM + c8, &sK[buf][row * KV_STRIDE + c8]);
      async_copy16(Vhead + (size_t)row * S_LEN + j0 + c8,     &sV[buf][row * KV_STRIDE + c8]);
    }
  };

  prefetch(0, 0);

  for (int it = 0; it < S_LEN / 64; ++it) {
    const int j0  = it * 64;
    const int cur = it & 1;
    wait_async_all();        // tile `cur` landed in LDS
    __syncthreads();         // visible to all waves; prev reads of buf cur^1 done
    if (it + 1 < S_LEN / 64) prefetch(cur ^ 1, j0 + 64);   // overlap with compute
    const __bf16* Kt = sK[cur];
    const __bf16* Vc = sV[cur];

    // ---- scores: 4 tiles of 16x16, each = Q(16x64) . K^T (Q pre-scaled) ----
    v8f sc[4];
#pragma unroll
    for (int jt = 0; jt < 4; ++jt) {
      v8f acc;
#pragma unroll
      for (int r = 0; r < 8; ++r) acc[r] = 0.f;
#pragma unroll
      for (int kf = 0; kf < 2; ++kf) {
        v16bf kb = lds_load32(Kt + (jt * 16 + nn) * KV_STRIDE + kf * 32 + g * 16);
        acc = __builtin_amdgcn_wmma_f32_16x16x32_bf16(false, qa[kf], false, kb, (short)0, acc, false, false);
      }
      sc[jt] = acc;
    }

    // ---- online softmax (row m = g*8+r across 16 lanes, n = lane&15) ----
    float mnew[8], alpha[8];
#pragma unroll
    for (int r = 0; r < 8; ++r) {
      float v = fmaxf(fmaxf(sc[0][r], sc[1][r]), fmaxf(sc[2][r], sc[3][r]));
#pragma unroll
      for (int msk = 1; msk < 16; msk <<= 1) v = fmaxf(v, __shfl_xor(v, msk, 32));
      mnew[r] = fmaxf(mrow[r], v);
      alpha[r] = __expf(mrow[r] - mnew[r]);
      mrow[r] = mnew[r];
    }
#pragma unroll
    for (int jt = 0; jt < 4; ++jt)
#pragma unroll
      for (int r = 0; r < 8; ++r) sc[jt][r] = __expf(sc[jt][r] - mnew[r]);
#pragma unroll
    for (int r = 0; r < 8; ++r) {
      float t = sc[0][r] + sc[1][r] + sc[2][r] + sc[3][r];
#pragma unroll
      for (int msk = 1; msk < 16; msk <<= 1) t += __shfl_xor(t, msk, 32);
      lrow[r] = lrow[r] * alpha[r] + t;
    }
#pragma unroll
    for (int dt = 0; dt < 4; ++dt)
#pragma unroll
      for (int r = 0; r < 8; ++r) o[dt][r] *= alpha[r];

    // ---- P (C-layout) -> per-wave LDS bounce -> A-fragment layout ----
#pragma unroll
    for (int jt = 0; jt < 4; ++jt)
#pragma unroll
      for (int r = 0; r < 8; ++r)
        ldsP[(g * 8 + r) * KV_STRIDE + jt * 16 + nn] = (__bf16)sc[jt][r];
    asm volatile("s_wait_dscnt 0" ::: "memory");

    v16bf pa[2];
    {
      const __bf16* prow = ldsP + nn * KV_STRIDE;
#pragma unroll
      for (int f = 0; f < 2; ++f)
#pragma unroll
        for (int c = 0; c < 2; ++c) {
          bf16x8 t = *(const bf16x8*)(prow + f * 32 + c * 8 + g * 16);
#pragma unroll
          for (int e = 0; e < 8; ++e) pa[f][c * 8 + e] = t[e];
        }
    }

    // ---- O += P(16x64) . V(64x64), V tile already [d][j'] in LDS ----
#pragma unroll
    for (int dt = 0; dt < 4; ++dt) {
#pragma unroll
      for (int kf = 0; kf < 2; ++kf) {
        v16bf vb = lds_load32(Vc + (dt * 16 + nn) * KV_STRIDE + kf * 32 + g * 16);
        o[dt] = __builtin_amdgcn_wmma_f32_16x16x32_bf16(false, pa[kf], false, vb, (short)0, o[dt], false, false);
      }
    }
  }

  // ---- epilogue: normalize and store to [B,S,H,64] f32 ----
  const int b = bh >> 3, h = bh & 7;
#pragma unroll
  for (int r = 0; r < 8; ++r) {
    float inv = 1.0f / lrow[r];
    int srow = q0 + g * 8 + r;
    size_t base = ((size_t)(b * S_LEN + srow) * NHEAD + h) * HEADDIM;
#pragma unroll
    for (int dt = 0; dt < 4; ++dt)
      O[base + dt * 16 + nn] = o[dt][r] * inv;
  }
}

// ---------------------------------------------------------------------------
// Kernel 3: residual add + LayerNorm over D=512.  One wave per token row.
// ---------------------------------------------------------------------------
__global__ __launch_bounds__(256) void resid_ln_kernel(
    const float* __restrict__ Obuf, const float* __restrict__ y,
    const float* __restrict__ gamma, const float* __restrict__ beta,
    float* __restrict__ out) {
  const int lane = threadIdx.x & 31;
  const int wave = threadIdx.x >> 5;
  const int row = blockIdx.x * 8 + wave;     // 0..8191
  const float* op = Obuf + (size_t)row * D_MODEL;
  const float* yp = y    + (size_t)row * D_MODEL;

  float v[16];
  float sum = 0.f;
#pragma unroll
  for (int j = 0; j < 16; ++j) {
    int d = j * 32 + lane;
    v[j] = op[d] + yp[d];
    sum += v[j];
  }
#pragma unroll
  for (int msk = 1; msk < 32; msk <<= 1) sum += __shfl_xor(sum, msk, 32);
  float mu = sum * (1.0f / 512.0f);
  float s2 = 0.f;
#pragma unroll
  for (int j = 0; j < 16; ++j) { float d = v[j] - mu; s2 += d * d; }
#pragma unroll
  for (int msk = 1; msk < 32; msk <<= 1) s2 += __shfl_xor(s2, msk, 32);
  float inv = rsqrtf(s2 * (1.0f / 512.0f) + 1e-5f);

  float* po = out + (size_t)row * D_MODEL;
#pragma unroll
  for (int j = 0; j < 16; ++j) {
    int d = j * 32 + lane;
    po[d] = (v[j] - mu) * inv * gamma[d] + beta[d];
  }
}

// ---------------------------------------------------------------------------
extern "C" void kernel_launch(void* const* d_in, const int* in_sizes, int n_in,
                              void* d_out, int out_size, void* d_ws, size_t ws_size,
                              hipStream_t stream) {
  const float* x     = (const float*)d_in[0];
  const float* y     = (const float*)d_in[1];
  const float* Wk    = (const float*)d_in[2];
  const float* bk    = (const float*)d_in[3];
  const float* Wq    = (const float*)d_in[4];
  const float* bq    = (const float*)d_in[5];
  const float* Wv    = (const float*)d_in[6];
  const float* bv    = (const float*)d_in[7];
  const float* gamma = (const float*)d_in[8];
  const float* beta  = (const float*)d_in[9];

  char* ws = (char*)d_ws;
  // 2*8*4096*64 bf16 = 8 MB each
  __bf16* Qb = (__bf16*)(ws);
  __bf16* Kb = (__bf16*)(ws + (8ull << 20));
  __bf16* Vt = (__bf16*)(ws + (16ull << 20));
  float*  Ob = (float*)(ws + (24ull << 20));   // 16 MB f32 [B,S,H,64]

  qkv_proj_kernel<<<1024, 128, 0, stream>>>(x, y, Wk, bk, Wq, bq, Wv, bv, Qb, Kb, Vt);
  flash_attn_kernel<<<dim3(64, 16), 128, 0, stream>>>(Qb, Kb, Vt, Ob);
  resid_ln_kernel<<<1024, 256, 0, stream>>>(Ob, y, gamma, beta, (float*)d_out);
}
